// CensorNet_50809463112199
// MI455X (gfx1250) — 8-bit matrix operand order repaired and checked by reading, compiled, not run on hardware
//
#include <hip/hip_runtime.h>

typedef __attribute__((ext_vector_type(8))) int   v8i;
typedef __attribute__((ext_vector_type(8))) float v8f;

#define TSTEPS 511
#define BDIM   256
#define IDIM   128
#define HDIM   256

__device__ __forceinline__ unsigned int f32_to_e4m3_sw(float f) {
  union { float fl; unsigned u; } v; v.fl = f;
  unsigned s = (v.u >> 31) << 7;
  float a = __builtin_fabsf(f);
  if (!(a < 448.0f)) return s | 0x7Eu;
  if (a < 0.015625f) {
    unsigned m = (unsigned)(a * 512.0f + 0.5f);
    return s | m;
  }
  unsigned e   = (v.u >> 23) & 0xFFu;
  unsigned man = v.u & 0x7FFFFFu;
  unsigned lsb = (man >> 20) & 1u;
  man += 0x7FFFFu + lsb;
  if (man >> 23) { man = 0u; e += 1u; }
  unsigned e4 = e - 120u;
  if (e4 > 15u || (e4 == 15u && (man >> 20) == 7u)) return s | 0x7Eu;
  return s | (e4 << 3) | (man >> 20);
}

__device__ __forceinline__ unsigned int pack2_e4m3(float a, float b) {
#if __has_builtin(__builtin_amdgcn_cvt_pk_fp8_f32)
  return (unsigned)__builtin_amdgcn_cvt_pk_fp8_f32(a, b, 0, false) & 0xFFFFu;
#else
  return f32_to_e4m3_sw(a) | (f32_to_e4m3_sw(b) << 8);
#endif
}
__device__ __forceinline__ unsigned int pack4_e4m3(const float* p) {
  return pack2_e4m3(p[0], p[1]) | (pack2_e4m3(p[2], p[3]) << 16);
}

__device__ __forceinline__ v8i load_afrag(const unsigned char* Mlds, int rowstride,
                                          int kbase, int lane) {
  const int m  = lane & 15;
  const int hi = (lane >> 4) & 1;
  const unsigned int* p = (const unsigned int*)(Mlds + m * rowstride);
  v8i a;
#pragma unroll
  for (int v = 0; v < 8; ++v) {
    int koff = kbase + (v & 3) * 4 + (v >> 2) * 32 + hi * 16;
    a[v] = (int)p[koff >> 2];
  }
  return a;
}

__device__ __forceinline__ v8i load_bfrag(const unsigned char* Wlds, int row,
                                          int rowstride, int kbase, int lane) {
  const int hi = (lane >> 4) & 1;
  const unsigned int* p = (const unsigned int*)(Wlds + row * rowstride);
  v8i b;
#pragma unroll
  for (int v = 0; v < 8; ++v) {
    int koff = kbase + (v & 3) * 4 + (v >> 2) * 32 + hi * 16;
    b[v] = (int)p[koff >> 2];
  }
  return b;
}

#define WMMA8(A, B, C) \
  __builtin_amdgcn_wmma_f32_16x16x64_fp8_fp8((A), (B), (short)0, (C), false, false)

#define SMEM_BYTES (768 * 128 + 768 * 256 + 16 * 128 + 16 * 256 + 64)

__global__ __launch_bounds__(256, 1) void gru_persistent_fp8(
    const float* __restrict__ x, const float* __restrict__ gtp,
    const float* __restrict__ W_ih, const float* __restrict__ W_hh,
    const float* __restrict__ b_ih, const float* __restrict__ b_hh,
    const float* __restrict__ W_dec, const float* __restrict__ b_dec,
    float* __restrict__ out) {
  extern __shared__ unsigned char smem[];
  unsigned char* Wih8 = smem;
  unsigned char* Whh8 = Wih8 + 768 * 128;
  unsigned char* Xf8  = Whh8 + 768 * 256;
  unsigned char* Hf8  = Xf8 + 16 * 128;
  float* logits       = (float*)(Hf8 + 16 * 256);

  const int tid    = threadIdx.x;
  const int lane   = tid & 31;
  const int w      = tid >> 5;
  const int hi     = (lane >> 4) & 1;
  const int n      = lane & 15;
  const int wgbase = blockIdx.x * 16;

  for (int i = tid * 4; i < 768 * 128; i += 256 * 4) {
    float q[4] = {W_ih[i], W_ih[i + 1], W_ih[i + 2], W_ih[i + 3]};
    *(unsigned int*)(Wih8 + i) = pack4_e4m3(q);
  }
  for (int i = tid * 4; i < 768 * 256; i += 256 * 4) {
    float q[4] = {W_hh[i], W_hh[i + 1], W_hh[i + 2], W_hh[i + 3]};
    *(unsigned int*)(Whh8 + i) = pack4_e4m3(q);
  }
  for (int i = tid * 4; i < 16 * 256; i += 256 * 4)
    *(unsigned int*)(Hf8 + i) = 0u;
  if (tid < 16) logits[tid] = 0.0f;

  float biasr[2], biasz[2], biasni[2], biasnh[2], wdec[2];
#pragma unroll
  for (int t2 = 0; t2 < 2; ++t2) {
    int j = w * 32 + t2 * 16 + n;
    biasr[t2]  = b_ih[j] + b_hh[j];
    biasz[t2]  = b_ih[256 + j] + b_hh[256 + j];
    biasni[t2] = b_ih[512 + j];
    biasnh[t2] = b_hh[512 + j];
    wdec[t2]   = W_dec[j];
  }
  const float bdec = b_dec[0];

  float hreg[2][8];
#pragma unroll
  for (int t2 = 0; t2 < 2; ++t2)
#pragma unroll
    for (int r = 0; r < 8; ++r) hreg[t2][r] = 0.0f;
  float loss_acc = 0.0f;

  __syncthreads();

  for (int t = 0; t < TSTEPS; ++t) {
    const float* xsrc = x + ((size_t)t * BDIM + wgbase) * IDIM + tid * 8;
    {
      float q0[4] = {xsrc[0], xsrc[1], xsrc[2], xsrc[3]};
      float q1[4] = {xsrc[4], xsrc[5], xsrc[6], xsrc[7]};
      unsigned int* dst = (unsigned int*)(Xf8 + tid * 8);
      dst[0] = pack4_e4m3(q0);
      dst[1] = pack4_e4m3(q1);
    }
    __syncthreads();

    v8i ax0 = load_afrag(Xf8, 128, 0, lane);
    v8i ax1 = load_afrag(Xf8, 128, 64, lane);
    v8i ah0 = load_afrag(Hf8, 256, 0, lane);
    v8i ah1 = load_afrag(Hf8, 256, 64, lane);
    v8i ah2 = load_afrag(Hf8, 256, 128, lane);
    v8i ah3 = load_afrag(Hf8, 256, 192, lane);

    float part[8];
#pragma unroll
    for (int r = 0; r < 8; ++r) part[r] = 0.0f;

#pragma unroll
    for (int t2 = 0; t2 < 2; ++t2) {
      const int brow = w * 32 + t2 * 16 + n;
      v8f ar, az, ani, anh;
#pragma unroll
      for (int i = 0; i < 8; ++i) {
        ar[i] = biasr[t2]; az[i] = biasz[t2]; ani[i] = biasni[t2]; anh[i] = biasnh[t2];
      }
      ar  = WMMA8(ax0, load_bfrag(Wih8, brow,        128, 0,  lane), ar);
      ar  = WMMA8(ax1, load_bfrag(Wih8, brow,        128, 64, lane), ar);
      az  = WMMA8(ax0, load_bfrag(Wih8, 256 + brow,  128, 0,  lane), az);
      az  = WMMA8(ax1, load_bfrag(Wih8, 256 + brow,  128, 64, lane), az);
      ani = WMMA8(ax0, load_bfrag(Wih8, 512 + brow,  128, 0,  lane), ani);
      ani = WMMA8(ax1, load_bfrag(Wih8, 512 + brow,  128, 64, lane), ani);
      ar  = WMMA8(ah0, load_bfrag(Whh8, brow,        256, 0,   lane), ar);
      ar  = WMMA8(ah1, load_bfrag(Whh8, brow,        256, 64,  lane), ar);
      ar  = WMMA8(ah2, load_bfrag(Whh8, brow,        256, 128, lane), ar);
      ar  = WMMA8(ah3, load_bfrag(Whh8, brow,        256, 192, lane), ar);
      az  = WMMA8(ah0, load_bfrag(Whh8, 256 + brow,  256, 0,   lane), az);
      az  = WMMA8(ah1, load_bfrag(Whh8, 256 + brow,  256, 64,  lane), az);
      az  = WMMA8(ah2, load_bfrag(Whh8, 256 + brow,  256, 128, lane), az);
      az  = WMMA8(ah3, load_bfrag(Whh8, 256 + brow,  256, 192, lane), az);
      anh = WMMA8(ah0, load_bfrag(Whh8, 512 + brow,  256, 0,   lane), anh);
      anh = WMMA8(ah1, load_bfrag(Whh8, 512 + brow,  256, 64,  lane), anh);
      anh = WMMA8(ah2, load_bfrag(Whh8, 512 + brow,  256, 128, lane), anh);
      anh = WMMA8(ah3, load_bfrag(Whh8, 512 + brow,  256, 192, lane), anh);

#pragma unroll
      for (int r = 0; r < 8; ++r) {
        float rr = 1.0f / (1.0f + __expf(-ar[r]));
        float zz = 1.0f / (1.0f + __expf(-az[r]));
        float nn = tanhf(ani[r] + rr * anh[r]);
        float h1 = (1.0f - zz) * nn + zz * hreg[t2][r];
        hreg[t2][r] = h1;
        part[r] += h1 * wdec[t2];
      }
    }
    __syncthreads();

#pragma unroll
    for (int t2 = 0; t2 < 2; ++t2) {
#pragma unroll
      for (int r = 0; r < 8; r += 2) {
        unsigned p2 = pack2_e4m3(hreg[t2][r], hreg[t2][r + 1]);
        int base = w * 32 + t2 * 16 + n;
        Hf8[(r + hi * 8) * 256 + base]     = (unsigned char)(p2 & 0xFF);
        Hf8[(r + 1 + hi * 8) * 256 + base] = (unsigned char)(p2 >> 8);
      }
    }

#pragma unroll
    for (int r = 0; r < 8; ++r) {
      float v = part[r];
      v += __shfl_xor(v, 1, 32);
      v += __shfl_xor(v, 2, 32);
      v += __shfl_xor(v, 4, 32);
      v += __shfl_xor(v, 8, 32);
      if (n == 0) atomicAdd(&logits[r + hi * 8], v);
    }
    __syncthreads();

    if (tid < 16) {
      float logit = logits[tid] + bdec;
      float g = gtp[(size_t)(t + 1) * BDIM + wgbase + tid];
      float lsp = (logit >= 0.0f) ? -log1pf(__expf(-logit))
                                  : (logit - log1pf(__expf(logit)));
      float lsn = (logit <= 0.0f) ? -log1pf(__expf(logit))
                                  : (-logit - log1pf(__expf(-logit)));
      loss_acc -= g * lsp + (1.0f - g) * lsn;
      logits[tid] = 0.0f;
    }
  }

  if (tid < 16) atomicAdd(out, loss_acc);
}

extern "C" void kernel_launch(void* const* d_in, const int* in_sizes, int n_in,
                              void* d_out, int out_size, void* d_ws, size_t ws_size,
                              hipStream_t stream) {
  (void)in_sizes; (void)n_in; (void)out_size; (void)d_ws; (void)ws_size;
  const float* x     = (const float*)d_in[0];
  const float* gtp   = (const float*)d_in[1];
  const float* W_ih  = (const float*)d_in[2];
  const float* W_hh  = (const float*)d_in[3];
  const float* b_ih  = (const float*)d_in[4];
  const float* b_hh  = (const float*)d_in[5];
  const float* W_dec = (const float*)d_in[6];
  const float* b_dec = (const float*)d_in[7];
  float* out = (float*)d_out;

  (void)hipFuncSetAttribute((const void*)gru_persistent_fp8,
                            hipFuncAttributeMaxDynamicSharedMemorySize, SMEM_BYTES);
  (void)hipMemsetAsync(d_out, 0, sizeof(float), stream);
  gru_persistent_fp8<<<BDIM / 16, 256, SMEM_BYTES, stream>>>(
      x, gtp, W_ih, W_hh, b_ih, b_hh, W_dec, b_dec, out);
}
